// Decoder_9225589752361
// MI455X (gfx1250) — matrix operand order repaired and checked by reading, not compiled, not run on hardware
//
#include <hip/hip_runtime.h>
#include <math.h>

// ---------------------------------------------------------------------------
// Persistent 2-layer LSTM decoder for MI455X (gfx1250, wave32, WMMA).
// One workgroup owns a 16-row batch tile and iterates all T timesteps.
// Weights converted to bf16 once into LDS; GEMMs use v_wmma_f32_16x16x32_bf16.
// Elementwise gates use branch-free __expf-based sigmoid/tanh to keep the
// serial per-timestep critical path short (no libm slow paths / divergence).
// ---------------------------------------------------------------------------

typedef __attribute__((ext_vector_type(16))) __bf16 v16bf;
typedef __attribute__((ext_vector_type(8)))  __bf16 v8bf;
typedef __attribute__((ext_vector_type(8)))  float  v8f;

static constexpr int Bsz   = 512;
static constexpr int Tlen  = 1024;
static constexpr int Hd    = 64;
static constexpr int Vocab = 28;
static constexpr int Gn    = 4 * Hd;      // 256 gate columns

// LDS layout (bytes). All offsets 16B-aligned. Strides padded (bf16 elems)
// so the 16 per-lane row addresses of a fragment land on distinct banks.
static constexpr int S_W0  = 200;         // W0: [256][192] K, stride 200
static constexpr int S_W1  = 136;         // W1: [256][128] K, stride 136
static constexpr int S_HW  = 72;          // head: [32][64] K, stride 72
static constexpr int S_XS0 = 200;         // x-stage L0: [16][192]
static constexpr int S_XS1 = 136;         // x-stage L1: [16][128]
static constexpr int S_GT  = 260;         // gates f32: [16][256]

static constexpr int OFF_W0  = 0;                              // 256*200*2 = 102400
static constexpr int OFF_W1  = OFF_W0  + 256 * S_W0 * 2;       // 102400
static constexpr int OFF_HW  = OFF_W1  + 256 * S_W1 * 2;       // 172032
static constexpr int OFF_EMB = OFF_HW  + 32 * S_HW * 2;        // 176640 ([28][64] bf16)
static constexpr int OFF_XS0 = OFF_EMB + 28 * Hd * 2;          // 180224
static constexpr int OFF_XS1 = OFF_XS0 + 16 * S_XS0 * 2;       // 186624
static constexpr int OFF_GT  = OFF_XS1 + 16 * S_XS1 * 2;       // 190976
static constexpr int OFF_B0  = OFF_GT  + 16 * S_GT * 4;        // 207616
static constexpr int OFF_B1  = OFF_B0  + Gn * 4;               // 208640
static constexpr int OFF_HB  = OFF_B1  + Gn * 4;               // 209664
static constexpr size_t SMEM_BYTES = OFF_HB + 32 * 4;          // 209792

// Branch-free activations on v_exp_f32; clamp keeps exp finite (exact
// saturation anyway for |x|>15 in fp32).
__device__ __forceinline__ float sigmoidf_(float x) {
    x = __builtin_fmaxf(__builtin_fminf(x, 30.0f), -30.0f);
    return 1.0f / (1.0f + __expf(-x));
}
__device__ __forceinline__ float tanhf_(float x) {
    x = __builtin_fmaxf(__builtin_fminf(x, 15.0f), -15.0f);
    const float e = __expf(2.0f * x);
    return (e - 1.0f) / (e + 1.0f);
}

// A fragment (16x32 bf16, M x K), ISA layout: lane<16 holds M=lane,
// K = {kb..kb+7, kb+16..kb+23}; lane>=16 holds M=lane-16, K shifted by 8.
__device__ __forceinline__ v16bf ldA(const __bf16* base, int stride, int lane, int kb) {
    const __bf16* p = base + (lane & 15) * stride + kb + ((lane & 16) ? 8 : 0);
    v8bf lo = *(const v8bf*)p;
    v8bf hi = *(const v8bf*)(p + 16);
    return __builtin_shufflevector(lo, hi, 0,1,2,3,4,5,6,7,8,9,10,11,12,13,14,15);
}

// B fragment (32x16 bf16, K x N) with B[k][n] = W[n][k] (W row-major [N][K]):
// lane holds column n = lane&15, K = {kb..kb+7, kb+16..kb+23} for lane<16,
// K shifted by 8 for lane>=16 (same K order as the A fragment).
__device__ __forceinline__ v16bf ldB(const __bf16* Wtile, int stride, int lane, int kb) {
    const __bf16* p = Wtile + (lane & 15) * stride + kb + ((lane & 16) ? 8 : 0);
    v8bf lo = *(const v8bf*)p;
    v8bf hi = *(const v8bf*)(p + 16);
    return __builtin_shufflevector(lo, hi, 0,1,2,3,4,5,6,7,8,9,10,11,12,13,14,15);
}

__device__ __forceinline__ v8f wmma_bf16(v16bf a, v16bf b, v8f c) {
    return __builtin_amdgcn_wmma_f32_16x16x32_bf16(
        /*neg_a=*/false, a, /*neg_b=*/false, b,
        /*c_mod=*/(short)0, c, /*reuse_a=*/false, /*reuse_b=*/false);
}

__global__ __launch_bounds__(256, 1)
void lstm_decoder_persistent(
    const int*   __restrict__ seq,       // [B,T]
    const float* __restrict__ h0in,      // [2,B,H]
    const float* __restrict__ c0in,      // [2,B,H]
    const float* __restrict__ enc,       // [B,T,H]
    const float* __restrict__ embf,      // [V,H]
    const float* __restrict__ w_ih0,     // [256,128]
    const float* __restrict__ w_hh0,     // [256,64]
    const float* __restrict__ b_ih0, const float* __restrict__ b_hh0,
    const float* __restrict__ w_ih1,     // [256,64]
    const float* __restrict__ w_hh1,     // [256,64]
    const float* __restrict__ b_ih1, const float* __restrict__ b_hh1,
    const float* __restrict__ head_w,    // [28,64]
    const float* __restrict__ head_b,    // [28]
    float* __restrict__ out)             // logits [B,T,V] ++ hT[2,B,H] ++ cT[2,B,H]
{
    extern __shared__ char smem[];
    __bf16* W0  = (__bf16*)(smem + OFF_W0);
    __bf16* W1  = (__bf16*)(smem + OFF_W1);
    __bf16* HW  = (__bf16*)(smem + OFF_HW);
    __bf16* EMB = (__bf16*)(smem + OFF_EMB);
    __bf16* XS0 = (__bf16*)(smem + OFF_XS0);
    __bf16* XS1 = (__bf16*)(smem + OFF_XS1);
    float*  GT  = (float*)(smem + OFF_GT);
    float*  B0s = (float*)(smem + OFF_B0);
    float*  B1s = (float*)(smem + OFF_B1);
    float*  HBs = (float*)(smem + OFF_HB);

    const int tid     = threadIdx.x;
    const int lane    = tid & 31;
    const int w       = tid >> 5;          // wave id 0..7
    const int col     = lane & 15;         // N within tile (also C-frag column)
    const int rowbase = blockIdx.x * 16;   // batch tile base

    float* outH = out + (size_t)Bsz * Tlen * Vocab;   // hT [2,B,H]
    float* outC = outH + 2 * Bsz * Hd;                // cT [2,B,H]

    // ---- one-time: stage bf16 weights / biases into LDS --------------------
    for (int i = tid; i < 256 * 128; i += 256)
        W0[(i >> 7) * S_W0 + (i & 127)] = (__bf16)w_ih0[i];
    for (int i = tid; i < 256 * 64; i += 256)
        W0[(i >> 6) * S_W0 + 128 + (i & 63)] = (__bf16)w_hh0[i];
    for (int i = tid; i < 256 * 64; i += 256)
        W1[(i >> 6) * S_W1 + (i & 63)] = (__bf16)w_ih1[i];
    for (int i = tid; i < 256 * 64; i += 256)
        W1[(i >> 6) * S_W1 + 64 + (i & 63)] = (__bf16)w_hh1[i];
    for (int i = tid; i < 32 * 64; i += 256) {
        int r = i >> 6, c = i & 63;
        HW[r * S_HW + c] = (r < Vocab) ? (__bf16)head_w[r * Hd + c] : (__bf16)0.0f;
    }
    for (int i = tid; i < Vocab * Hd; i += 256) EMB[i] = (__bf16)embf[i];
    B0s[tid] = b_ih0[tid] + b_hh0[tid];
    B1s[tid] = b_ih1[tid] + b_hh1[tid];
    if (tid < 32) HBs[tid] = (tid < Vocab) ? head_b[tid] : 0.0f;

    // ---- per-thread recurrent state: c of 4 hidden units, both layers ------
    const int em = tid >> 4;          // batch row within tile (0..15)
    const int eh = (tid & 15) * 4;    // hidden base (0..60)
    const int eb = rowbase + em;
    float cr0[4], cr1[4];
#pragma unroll
    for (int j = 0; j < 4; ++j) {
        cr0[j] = c0in[(size_t)eb * Hd + eh + j];
        cr1[j] = c0in[(size_t)(Bsz + eb) * Hd + eh + j];
        XS0[em * S_XS0 + 128 + eh + j] = (__bf16)h0in[(size_t)eb * Hd + eh + j];
        XS1[em * S_XS1 + 64 + eh + j]  = (__bf16)h0in[(size_t)(Bsz + eb) * Hd + eh + j];
    }
    __syncthreads();

    for (int t = 0; t < Tlen; ++t) {
        // -- phase a: stage encoder (cols 0..63) + embedding (64..127) -------
        {
            const size_t erow = ((size_t)eb * Tlen + t) * Hd;
            const float4 ev = *(const float4*)(enc + erow + (tid & 15) * 4);
            __bf16* xr = XS0 + em * S_XS0 + (tid & 15) * 4;
            xr[0] = (__bf16)ev.x; xr[1] = (__bf16)ev.y;
            xr[2] = (__bf16)ev.z; xr[3] = (__bf16)ev.w;
            const int id = seq[(size_t)eb * Tlen + t];
            __bf16* xe = xr + 64;
            if (id == 0) {
#pragma unroll
                for (int j = 0; j < 4; ++j) xe[j] = (__bf16)0.0f;
            } else {
                const __bf16* er = EMB + id * Hd + (tid & 15) * 4;
#pragma unroll
                for (int j = 0; j < 4; ++j) xe[j] = er[j];
            }
            if (t + 1 < Tlen)   // gfx1250 global_prefetch_b8 for next step
                __builtin_prefetch(enc + erow + Hd + (tid & 15) * 4, 0, 0);
        }
        __syncthreads();

        // -- phase b: layer-0 GEMM  [16x192] @ [192x256] ---------------------
        {
            const __bf16* Wt0 = W0 + (2 * w) * 16 * S_W0;
            const __bf16* Wt1 = W0 + (2 * w + 1) * 16 * S_W0;
            v8f a0, a1;
            const float bi0 = B0s[(2 * w) * 16 + col];
            const float bi1 = B0s[(2 * w + 1) * 16 + col];
#pragma unroll
            for (int r = 0; r < 8; ++r) { a0[r] = bi0; a1[r] = bi1; }
#pragma unroll
            for (int kc = 0; kc < 6; ++kc) {
                v16bf A = ldA(XS0, S_XS0, lane, kc * 32);
                a0 = wmma_bf16(A, ldB(Wt0, S_W0, lane, kc * 32), a0);
                a1 = wmma_bf16(A, ldB(Wt1, S_W0, lane, kc * 32), a1);
            }
            const int Mb = (lane & 16) ? 8 : 0;
#pragma unroll
            for (int r = 0; r < 8; ++r) {
                GT[(Mb + r) * S_GT + (2 * w) * 16 + col]     = a0[r];
                GT[(Mb + r) * S_GT + (2 * w + 1) * 16 + col] = a1[r];
            }
        }
        __syncthreads();

        // -- phase c: layer-0 gates -> h0n, c0n ------------------------------
        {
            const float* gr = GT + em * S_GT;
#pragma unroll
            for (int j = 0; j < 4; ++j) {
                const int h = eh + j;
                const float ig = sigmoidf_(gr[h]);
                const float fg = sigmoidf_(gr[64 + h]);
                const float gg = tanhf_(gr[128 + h]);
                const float og = sigmoidf_(gr[192 + h]);
                const float cn = fg * cr0[j] + ig * gg;
                const float hn = og * tanhf_(cn);
                cr0[j] = cn;
                XS0[em * S_XS0 + 128 + h] = (__bf16)hn;
                XS1[em * S_XS1 + h]       = (__bf16)hn;
                if (t == Tlen - 1) {
                    outH[(size_t)eb * Hd + h] = hn;
                    outC[(size_t)eb * Hd + h] = cn;
                }
            }
        }
        __syncthreads();

        // -- phase d: layer-1 GEMM  [16x128] @ [128x256] ---------------------
        {
            const __bf16* Wt0 = W1 + (2 * w) * 16 * S_W1;
            const __bf16* Wt1 = W1 + (2 * w + 1) * 16 * S_W1;
            v8f a0, a1;
            const float bi0 = B1s[(2 * w) * 16 + col];
            const float bi1 = B1s[(2 * w + 1) * 16 + col];
#pragma unroll
            for (int r = 0; r < 8; ++r) { a0[r] = bi0; a1[r] = bi1; }
#pragma unroll
            for (int kc = 0; kc < 4; ++kc) {
                v16bf A = ldA(XS1, S_XS1, lane, kc * 32);
                a0 = wmma_bf16(A, ldB(Wt0, S_W1, lane, kc * 32), a0);
                a1 = wmma_bf16(A, ldB(Wt1, S_W1, lane, kc * 32), a1);
            }
            const int Mb = (lane & 16) ? 8 : 0;
#pragma unroll
            for (int r = 0; r < 8; ++r) {
                GT[(Mb + r) * S_GT + (2 * w) * 16 + col]     = a0[r];
                GT[(Mb + r) * S_GT + (2 * w + 1) * 16 + col] = a1[r];
            }
        }
        __syncthreads();

        // -- phase e: layer-1 gates -> h1n, c1n ------------------------------
        {
            const float* gr = GT + em * S_GT;
#pragma unroll
            for (int j = 0; j < 4; ++j) {
                const int h = eh + j;
                const float ig = sigmoidf_(gr[h]);
                const float fg = sigmoidf_(gr[64 + h]);
                const float gg = tanhf_(gr[128 + h]);
                const float og = sigmoidf_(gr[192 + h]);
                const float cn = fg * cr1[j] + ig * gg;
                const float hn = og * tanhf_(cn);
                cr1[j] = cn;
                XS1[em * S_XS1 + 64 + h] = (__bf16)hn;
                if (t == Tlen - 1) {
                    outH[(size_t)(Bsz + eb) * Hd + h] = hn;
                    outC[(size_t)(Bsz + eb) * Hd + h] = cn;
                }
            }
        }
        __syncthreads();

        // -- phase f: head GEMM [16x64] @ [64x32pad] -> logits (waves 0,1) ---
        if (w < 2) {
            const __bf16* Wt = HW + w * 16 * S_HW;
            const int n = w * 16 + col;
            v8f acc;
            const float bb = (n < Vocab) ? HBs[n] : 0.0f;
#pragma unroll
            for (int r = 0; r < 8; ++r) acc[r] = bb;
#pragma unroll
            for (int kc = 0; kc < 2; ++kc) {
                v16bf A = ldA(XS1 + 64, S_XS1, lane, kc * 32);
                acc = wmma_bf16(A, ldB(Wt, S_HW, lane, kc * 32), acc);
            }
            if (n < Vocab) {
                const int Mb = (lane & 16) ? 8 : 0;
#pragma unroll
                for (int r = 0; r < 8; ++r) {
                    const int b = rowbase + Mb + r;
                    out[((size_t)b * Tlen + t) * Vocab + n] = acc[r];
                }
            }
        }
        // no trailing barrier needed: next iteration's phase-a writes touch
        // only XS0[:,0:128], disjoint from phase-f reads (XS1[:,64:128], HW).
    }
}

extern "C" void kernel_launch(void* const* d_in, const int* in_sizes, int n_in,
                              void* d_out, int out_size, void* d_ws, size_t ws_size,
                              hipStream_t stream) {
    (void)in_sizes; (void)n_in; (void)d_ws; (void)ws_size; (void)out_size;
    const int*   seq    = (const int*)  d_in[0];
    const float* h0     = (const float*)d_in[1];
    const float* c0     = (const float*)d_in[2];
    const float* enc    = (const float*)d_in[3];
    const float* embf   = (const float*)d_in[4];
    const float* w_ih0  = (const float*)d_in[5];
    const float* w_hh0  = (const float*)d_in[6];
    const float* b_ih0  = (const float*)d_in[7];
    const float* b_hh0  = (const float*)d_in[8];
    const float* w_ih1  = (const float*)d_in[9];
    const float* w_hh1  = (const float*)d_in[10];
    const float* b_ih1  = (const float*)d_in[11];
    const float* b_hh1  = (const float*)d_in[12];
    const float* head_w = (const float*)d_in[13];
    const float* head_b = (const float*)d_in[14];
    float* out = (float*)d_out;

    dim3 grid(Bsz / 16), block(256);
    lstm_decoder_persistent<<<grid, block, SMEM_BYTES, stream>>>(
        seq, h0, c0, enc, embf,
        w_ih0, w_hh0, b_ih0, b_hh0,
        w_ih1, w_hh1, b_ih1, b_hh1,
        head_w, head_b, out);
}
